// SelfAttention_84885733638743
// MI455X (gfx1250) — matrix operand order repaired and checked by reading, not compiled, not run on hardware
//
#include <hip/hip_runtime.h>
#include <hip/hip_bf16.h>

// ---------------------------------------------------------------------------
// MI455X (gfx1250) attention block, WMMA f16 pipeline, software-pipelined,
// with async global->LDS staging (ASYNCcnt) for the flash-attention K tiles.
//   B=16, C=512, H=W=32 (N=1024), heads=8, Dh=64, groups=8.
// All GEMMs use v_wmma_f32_16x16x32_f16 with NT data layouts so that both
// A and B fragments are contiguous 32-half K-runs (16B vector loads).
// ---------------------------------------------------------------------------

typedef __attribute__((ext_vector_type(16))) _Float16 v16h;
typedef __attribute__((ext_vector_type(8)))  _Float16 v8h;
typedef __attribute__((ext_vector_type(4)))  _Float16 v4h;
typedef __attribute__((ext_vector_type(8)))  float    v8f;

#define NB  16      // batch
#define NC  512     // channels
#define NN  1024    // tokens (H*W)
#define NHH 8       // heads
#define NDH 64      // head dim
#define LDP 72      // padded LDS row stride in halves (multiple of 8 -> 16B aligned)

// Build a v16h from two 16-byte chunks.
__device__ inline v16h ld16(const _Float16* p0, const _Float16* p1) {
  v8h lo = *(const v8h*)p0;
  v8h hi = *(const v8h*)p1;
  v16h r;
#pragma unroll
  for (int i = 0; i < 8; ++i) { r[i] = lo[i]; r[8 + i] = hi[i]; }
  return r;
}
// A fragment (16x32 MxK, row-major src): p = rowbase + sel*8 ; halves 0-7 = p[0..7],
// halves 8-15 = p[16..23]  (ISA 16-bit A layout: half sel holds K sel*8..+7, 16+sel*8..+7)
__device__ inline v16h ldA(const _Float16* p) { return ld16(p, p + 16); }
// B fragment (32x16 KxN from "N-major rows, K contiguous" src): p = colbase + sel*16 ;
// same K order as ldA: q = colbase + sel*8, halves 0-7 = q[0..7], halves 8-15 = q[16..23]
__device__ inline v16h ldB(const _Float16* p) {
  const _Float16* q = p - 8 * ((threadIdx.x & 31) >> 4);
  return ld16(q, q + 16);
}

__device__ inline v8f wmma16(v16h a, v16h b, v8f c) {
  return __builtin_amdgcn_wmma_f32_16x16x32_f16(false, a, false, b, (short)0, c,
                                                false, false);
}

// Async DMA: 16 bytes per lane, global -> LDS, tracked by ASYNCcnt.
// VDST = VGPR with LDS byte address, VADDR = VGPR pair with global address.
__device__ inline void async_ld16(_Float16* lds_dst, const _Float16* gsrc) {
  unsigned ldsa = (unsigned)(unsigned long long)(uintptr_t)lds_dst; // low 32b = LDS offset
  unsigned long long ga = (unsigned long long)(uintptr_t)gsrc;
  asm volatile("global_load_async_to_lds_b128 %0, %1, off"
               :: "v"(ldsa), "v"(ga) : "memory");
}

// ---------------------------------------------------------------------------
// Kernel 0: weight f32 -> f16
// ---------------------------------------------------------------------------
__global__ void wcvt_kernel(const float* __restrict__ src, _Float16* __restrict__ dst) {
  int i = blockIdx.x * 256 + threadIdx.x;   // i indexes float4 groups, n = 262144
  float4 v = ((const float4*)src)[i];
  v4h o;
  o[0] = (_Float16)v.x; o[1] = (_Float16)v.y; o[2] = (_Float16)v.z; o[3] = (_Float16)v.w;
  ((v4h*)dst)[i] = o;
}

// ---------------------------------------------------------------------------
// Kernel 1: GroupNorm -> xn_t[b, n, c] (f16, channel-contiguous rows)
// ---------------------------------------------------------------------------
__global__ __launch_bounds__(256)
void groupnorm_kernel(const float* __restrict__ x, const float* __restrict__ gamma,
                      const float* __restrict__ beta, _Float16* __restrict__ xn_t) {
  const int b = blockIdx.x >> 3;
  const int g = blockIdx.x & 7;
  const int tid = threadIdx.x;
  const float* xg = x + ((size_t)b * NC + g * 64) * NN;
  const float4* xg4 = (const float4*)xg;

  float s = 0.f, s2 = 0.f;
  for (int i = tid; i < 64 * NN / 4; i += 256) {
    float4 v = xg4[i];
    s  += v.x + v.y + v.z + v.w;
    s2 += v.x * v.x + v.y * v.y + v.z * v.z + v.w * v.w;
  }
  __shared__ float red0[256], red1[256];
  red0[tid] = s; red1[tid] = s2;
  __syncthreads();
  for (int off = 128; off > 0; off >>= 1) {
    if (tid < off) { red0[tid] += red0[tid + off]; red1[tid] += red1[tid + off]; }
    __syncthreads();
  }
  const float mean = red0[0] * (1.f / 65536.f);
  const float var  = red1[0] * (1.f / 65536.f) - mean * mean;
  const float inv  = rsqrtf(var + 1e-5f);

  for (int i = tid; i < 64 * NN / 4; i += 256) {
    float4 v = xg4[i];
    int base = i * 4;
    int c = g * 64 + (base >> 10);
    int n = base & (NN - 1);
    float gg = gamma[c] * inv;
    float bb = beta[c] - mean * gg;
    _Float16* d = xn_t + ((size_t)b * NN + n) * NC + c;
    d[0 * NC] = (_Float16)(v.x * gg + bb);
    d[1 * NC] = (_Float16)(v.y * gg + bb);
    d[2 * NC] = (_Float16)(v.z * gg + bb);
    d[3 * NC] = (_Float16)(v.w * gg + bb);
  }
}

// ---------------------------------------------------------------------------
// Kernel 2: QKV projection, NT GEMM, 128(n) x 64(o) tile, double-buffered.
// ---------------------------------------------------------------------------
__global__ __launch_bounds__(128)
void qkv_gemm_kernel(const _Float16* __restrict__ xn_t, const _Float16* __restrict__ wf16,
                     const float* __restrict__ bq, const float* __restrict__ bk,
                     const float* __restrict__ bv,
                     _Float16* __restrict__ q_t, _Float16* __restrict__ k_t,
                     _Float16* __restrict__ v_t) {
  const int n0   = blockIdx.x * 128;
  const int o0   = blockIdx.y * 64;
  const int b    = blockIdx.z / 3;
  const int proj = blockIdx.z % 3;
  const int wid  = threadIdx.x >> 5;
  const int lane = threadIdx.x & 31;
  const int r    = lane & 15;
  const int sel  = lane >> 4;

  const _Float16* W    = wf16 + (size_t)proj * NC * NC;
  const float*    bias = (proj == 0) ? bq : (proj == 1) ? bk : bv;
  _Float16*       dst  = (proj == 0) ? q_t : (proj == 1) ? k_t : v_t;
  const float     osc  = (proj == 0) ? 0.125f : 1.0f;   // Dh^-0.5 folded into q

  const _Float16* Arow0 = xn_t + ((size_t)b * NN + n0 + wid * 32 + r) * NC + sel * 8;
  const _Float16* Arow1 = Arow0 + 16 * NC;
  const _Float16* Bbase = W + (size_t)(o0 + r) * NC + sel * 16;

  v8f acc[2][4] = {};
  v16h a0[2], a1[2], bf[2][4];

  a0[0] = ldA(Arow0);
  a1[0] = ldA(Arow1);
#pragma unroll
  for (int dd = 0; dd < 4; ++dd) bf[0][dd] = ldB(Bbase + (size_t)dd * 16 * NC);

  for (int k0 = 0; k0 < NC; k0 += 32) {
    const int cur = (k0 >> 5) & 1, nxt = cur ^ 1;
    const int k1 = k0 + 32;
    if (k1 < NC) {                    // prefetch next-k fragments (in flight over WMMAs)
      a0[nxt] = ldA(Arow0 + k1);
      a1[nxt] = ldA(Arow1 + k1);
#pragma unroll
      for (int dd = 0; dd < 4; ++dd)
        bf[nxt][dd] = ldB(Bbase + (size_t)dd * 16 * NC + k1);
    }
#pragma unroll
    for (int dd = 0; dd < 4; ++dd) {
      acc[0][dd] = wmma16(a0[cur], bf[cur][dd], acc[0][dd]);
      acc[1][dd] = wmma16(a1[cur], bf[cur][dd], acc[1][dd]);
    }
  }

#pragma unroll
  for (int st = 0; st < 2; ++st) {
#pragma unroll
    for (int dd = 0; dd < 4; ++dd) {
      const int o = o0 + dd * 16 + r;
      const int h = o >> 6, d = o & 63;
      const float bo_ = bias[o];
#pragma unroll
      for (int e = 0; e < 8; ++e) {
        const int n = n0 + wid * 32 + st * 16 + e + 8 * sel;
        dst[(((size_t)b * NHH + h) * NN + n) * NDH + d] =
            (_Float16)((acc[st][dd][e] + bo_) * osc);
      }
    }
  }
}

// ---------------------------------------------------------------------------
// Kernel 3: flash attention, one (b,h) x 64-row block per workgroup, 4 waves.
// K tiles: async DMA global->LDS (double-buffered, ASYNCcnt pipelined).
// V tiles: register-staged and transposed into LDS (contiguous-K B frags).
// S = q kT; online softmax (shfl_xor row reductions); O += P V via LDS.
// ---------------------------------------------------------------------------
__global__ __launch_bounds__(128)
void flash_attn_kernel(const _Float16* __restrict__ q_t, const _Float16* __restrict__ k_t,
                       const _Float16* __restrict__ v_t, _Float16* __restrict__ att_t) {
  const int n0   = blockIdx.x * 64;
  const int bh   = blockIdx.y;             // b*8 + h
  const int wid  = threadIdx.x >> 5;
  const int lane = threadIdx.x & 31;
  const int r    = lane & 15;
  const int sel  = lane >> 4;

  const _Float16* qh = q_t + (size_t)bh * NN * NDH;
  const _Float16* kh = k_t + (size_t)bh * NN * NDH;
  const _Float16* vh = v_t + (size_t)bh * NN * NDH;

  __shared__ _Float16 k_lds [2][64 * LDP];   // double-buffered async K tiles
  __shared__ _Float16 vT_lds[64 * LDP];
  __shared__ _Float16 p_lds [4 * 16 * LDP];

  // persistent Q fragments: rows n0+wid*16 .. +15, K = d = 0..63
  const _Float16* qrow = qh + (size_t)(n0 + wid * 16 + r) * NDH + sel * 8;
  const v16h aq0 = ldA(qrow);
  const v16h aq1 = ldA(qrow + 32);

  // per-thread staging mapping (8-half chunks): i = tid*4+j -> row i>>3, chunk i&7
  const int srow   = (threadIdx.x * 4) >> 3;        // 1 row per thread (2 threads per row)
  const int schunk = (threadIdx.x * 4) & 7;         // starting chunk (0 or 4)
  v8h vreg[4];

  // prologue: async K tile 0 -> k_lds[0] (4 async b128 per wave); V tile 0 -> regs
#pragma unroll
  for (int j = 0; j < 4; ++j) {
    const int row = srow + ((schunk + j) >> 3), chunk = (schunk + j) & 7;
    async_ld16(&k_lds[0][row * LDP + chunk * 8], kh + (size_t)row * NDH + chunk * 8);
    vreg[j] = *(const v8h*)(vh + (size_t)row * NDH + chunk * 8);
  }

  v8f oacc[4] = {};
  float rmax[8], rsum[8];
#pragma unroll
  for (int e = 0; e < 8; ++e) { rmax[e] = -3.0e38f; rsum[e] = 0.f; }

  for (int m0 = 0; m0 < NN; m0 += 64) {
    const int cur = (m0 >> 6) & 1, nxt = cur ^ 1;
    __syncthreads();                 // all waves done reading vT_lds / k_lds[nxt]
    // commit V registers into LDS, transposed
#pragma unroll
    for (int j = 0; j < 4; ++j) {
      const int row = srow + ((schunk + j) >> 3), chunk = (schunk + j) & 7;
#pragma unroll
      for (int t = 0; t < 8; ++t) vT_lds[(chunk * 8 + t) * LDP + row] = vreg[j][t];
    }
    // launch next tile: async K straight into k_lds[nxt], V into registers
    if (m0 + 64 < NN) {
#pragma unroll
      for (int j = 0; j < 4; ++j) {
        const int row = srow + ((schunk + j) >> 3), chunk = (schunk + j) & 7;
        async_ld16(&k_lds[nxt][row * LDP + chunk * 8],
                   kh + (size_t)(m0 + 64 + row) * NDH + chunk * 8);
        vreg[j] = *(const v8h*)(vh + (size_t)(m0 + 64 + row) * NDH + chunk * 8);
      }
      // <=4 outstanding: the 4 just-issued; current tile's DMA is complete
      asm volatile("s_wait_asynccnt 0x4" ::: "memory");
    } else {
      asm volatile("s_wait_asynccnt 0x0" ::: "memory");
    }
    __syncthreads();                 // DMA'd K tile + vT_lds visible to all waves

    // S tile: 16 rows x 64 cols per wave
    v8f s[4] = {};
#pragma unroll
    for (int mm = 0; mm < 4; ++mm) {
      const _Float16* kr = &k_lds[cur][(mm * 16 + r) * LDP + sel * 16];
      s[mm] = wmma16(aq0, ldB(kr), s[mm]);
      s[mm] = wmma16(aq1, ldB(kr + 32), s[mm]);
    }

    // online softmax (row = e + 8*sel; its 16 cols live across lanes xor {1,2,4,8})
#pragma unroll
    for (int e = 0; e < 8; ++e) {
      float mloc = fmaxf(fmaxf(s[0][e], s[1][e]), fmaxf(s[2][e], s[3][e]));
#pragma unroll
      for (int mask = 1; mask < 16; mask <<= 1)
        mloc = fmaxf(mloc, __shfl_xor(mloc, mask, 32));
      const float mn = fmaxf(rmax[e], mloc);
      const float alpha = __expf(rmax[e] - mn);
      rmax[e] = mn;
#pragma unroll
      for (int dd = 0; dd < 4; ++dd) oacc[dd][e] *= alpha;
      float ps = 0.f;
#pragma unroll
      for (int mm = 0; mm < 4; ++mm) {
        const float p = __expf(s[mm][e] - mn);
        p_lds[wid * 16 * LDP + (e + 8 * sel) * LDP + mm * 16 + r] = (_Float16)p;
        ps += p;
      }
#pragma unroll
      for (int mask = 1; mask < 16; mask <<= 1)
        ps += __shfl_xor(ps, mask, 32);
      rsum[e] = rsum[e] * alpha + ps;
    }
    __syncthreads();

    // O += P * V   (A = P 16x64 from LDS, B = vT contiguous-K from LDS)
    const _Float16* pr = p_lds + wid * 16 * LDP + r * LDP;
    const v16h ap0 = ldA(pr + sel * 8);
    const v16h ap1 = ldA(pr + 32 + sel * 8);
#pragma unroll
    for (int dd = 0; dd < 4; ++dd) {
      const _Float16* vr = vT_lds + (dd * 16 + r) * LDP + sel * 16;
      oacc[dd] = wmma16(ap0, ldB(vr), oacc[dd]);
      oacc[dd] = wmma16(ap1, ldB(vr + 32), oacc[dd]);
    }
  }

  // epilogue: normalize by row sums, store att_t[bh, n, d]
#pragma unroll
  for (int dd = 0; dd < 4; ++dd) {
    const int d = dd * 16 + r;
#pragma unroll
    for (int e = 0; e < 8; ++e) {
      const int n = n0 + wid * 16 + e + 8 * sel;
      att_t[(size_t)bh * NN * NDH + (size_t)n * NDH + d] =
          (_Float16)(oacc[dd][e] / rsum[e]);
    }
  }
}

// ---------------------------------------------------------------------------
// Kernel 4: output projection + bias + residual, double-buffered like QKV.
// ---------------------------------------------------------------------------
__global__ __launch_bounds__(128)
void out_proj_kernel(const _Float16* __restrict__ att_t, const _Float16* __restrict__ wo16,
                     const float* __restrict__ bo, const float* __restrict__ x,
                     float* __restrict__ out) {
  const int n0   = blockIdx.x * 128;
  const int o0   = blockIdx.y * 64;
  const int b    = blockIdx.z;
  const int wid  = threadIdx.x >> 5;
  const int lane = threadIdx.x & 31;
  const int r    = lane & 15;
  const int sel  = lane >> 4;
  const int nA   = n0 + wid * 32 + r;        // strip 0 row; strip 1 = +16

  const _Float16* Bbase = wo16 + (size_t)(o0 + r) * NC + sel * 16;

  v8f acc[2][4] = {};
  v16h a0[2], a1[2], bf[2][4];

  {
    const _Float16* ar = att_t + ((size_t)b * NHH * NN + nA) * NDH + sel * 8;
    a0[0] = ldA(ar);
    a1[0] = ldA(ar + 16 * NDH);
#pragma unroll
    for (int dd = 0; dd < 4; ++dd) bf[0][dd] = ldB(Bbase + (size_t)dd * 16 * NC);
  }

  for (int k0 = 0; k0 < NC; k0 += 32) {
    const int cur = (k0 >> 5) & 1, nxt = cur ^ 1;
    const int k1 = k0 + 32;
    if (k1 < NC) {
      const int h = k1 >> 6, koff = k1 & 63;
      const _Float16* ar =
          att_t + (((size_t)b * NHH + h) * NN + nA) * NDH + koff + sel * 8;
      a0[nxt] = ldA(ar);
      a1[nxt] = ldA(ar + 16 * NDH);
#pragma unroll
      for (int dd = 0; dd < 4; ++dd)
        bf[nxt][dd] = ldB(Bbase + (size_t)dd * 16 * NC + k1);
    }
#pragma unroll
    for (int dd = 0; dd < 4; ++dd) {
      acc[0][dd] = wmma16(a0[cur], bf[cur][dd], acc[0][dd]);
      acc[1][dd] = wmma16(a1[cur], bf[cur][dd], acc[1][dd]);
    }
  }

#pragma unroll
  for (int st = 0; st < 2; ++st) {
#pragma unroll
    for (int dd = 0; dd < 4; ++dd) {
      const int o = o0 + dd * 16 + r;
      const float bb = bo[o];
#pragma unroll
      for (int e = 0; e < 8; ++e) {
        const int nn = n0 + wid * 32 + st * 16 + e + 8 * sel;
        const size_t idx = ((size_t)b * NC + o) * NN + nn;
        out[idx] = acc[st][dd][e] + bb + x[idx];
      }
    }
  }
}

// ---------------------------------------------------------------------------
// Host-side launcher
// ---------------------------------------------------------------------------
extern "C" void kernel_launch(void* const* d_in, const int* in_sizes, int n_in,
                              void* d_out, int out_size, void* d_ws, size_t ws_size,
                              hipStream_t stream) {
  const float* x     = (const float*)d_in[0];
  const float* wq    = (const float*)d_in[1];
  const float* bq    = (const float*)d_in[2];
  const float* wk    = (const float*)d_in[3];
  const float* bk    = (const float*)d_in[4];
  const float* wv    = (const float*)d_in[5];
  const float* bv    = (const float*)d_in[6];
  const float* wo    = (const float*)d_in[7];
  const float* bo    = (const float*)d_in[8];
  const float* gamma = (const float*)d_in[9];
  const float* beta  = (const float*)d_in[10];
  float* out = (float*)d_out;

  const size_t WSZ = (size_t)NC * NC;            // 262144
  const size_t ASZ = (size_t)NB * NN * NC;       // 8388608 (= B*Hh*N*Dh)
  _Float16* ws    = (_Float16*)d_ws;
  _Float16* wf16  = ws;                // 4 weight matrices, f16
  _Float16* xn_t  = wf16 + 4 * WSZ;    // [B, N, C]
  _Float16* q_t   = xn_t + ASZ;        // [B, Hh, N, Dh]
  _Float16* k_t   = q_t + ASZ;
  _Float16* v_t   = k_t + ASZ;
  _Float16* att_t = v_t + ASZ;         // [B, Hh, N, Dh]

  wcvt_kernel<<<256, 256, 0, stream>>>(wq, wf16 + 0 * WSZ);
  wcvt_kernel<<<256, 256, 0, stream>>>(wk, wf16 + 1 * WSZ);
  wcvt_kernel<<<256, 256, 0, stream>>>(wv, wf16 + 2 * WSZ);
  wcvt_kernel<<<256, 256, 0, stream>>>(wo, wf16 + 3 * WSZ);

  groupnorm_kernel<<<NB * 8, 256, 0, stream>>>(x, gamma, beta, xn_t);

  qkv_gemm_kernel<<<dim3(NN / 128, NC / 64, NB * 3), 128, 0, stream>>>(
      xn_t, wf16, bq, bk, bv, q_t, k_t, v_t);

  flash_attn_kernel<<<dim3(NN / 64, NB * NHH), 128, 0, stream>>>(q_t, k_t, v_t, att_t);

  out_proj_kernel<<<dim3(NN / 128, NC / 64, NB), 128, 0, stream>>>(
      att_t, wf16 + 3 * WSZ, bo, x, out);
}
